// CoLL_78065325572578
// MI455X (gfx1250) — compile-verified, matrix operand order checked by reading, not run on hardware
//
#include <hip/hip_runtime.h>
#include <hip/hip_bf16.h>
#include <stdint.h>

// ---------------------------------------------------------------------------
// Co-occurrence layer, collapsed form:
//   y(p) = sum_{9 taps} x_q * ( co[bin(x_p), bin(x_q)] * w_tap )
// Fused table T[64*9] computed on the WMMA unit; main kernel is an LDS-tiled
// 3x3 stencil with per-tap LUT gather, vectorized over channel quads.
// ---------------------------------------------------------------------------

#define NB    8
#define DIMN  8
#define DIMH  128
#define DIMW  128
#define DIMC  64
#define TH    8
#define TW    8
#define HALO_H (TH + 2)
#define HALO_W (TW + 2)
#define HALO_ELEMS (HALO_H * HALO_W * DIMC)   // 6400 floats

typedef __attribute__((ext_vector_type(2)))  float    v2f;
typedef __attribute__((ext_vector_type(8)))  float    v8f;
typedef __attribute__((ext_vector_type(16))) _Float16 v16h;
typedef __attribute__((ext_vector_type(4)))  int      v4i;

// ---- ordered-uint encoding for float min/max via integer atomics ----------
__device__ __forceinline__ unsigned enc_key(float f) {
  unsigned u = __float_as_uint(f);
  return (u & 0x80000000u) ? ~u : (u | 0x80000000u);
}
__device__ __forceinline__ float dec_key(unsigned k) {
  unsigned u = (k & 0x80000000u) ? (k ^ 0x80000000u) : ~k;
  return __uint_as_float(u);
}

// ---------------------------------------------------------------------------
// Kernel 1: init min/max keys in workspace
// ---------------------------------------------------------------------------
__global__ void coll_init_ws(unsigned* ws) {
  if (threadIdx.x == 0 && blockIdx.x == 0) {
    ws[0] = 0xFFFFFFFFu;  // min key identity
    ws[1] = 0x00000000u;  // max key identity
  }
}

// ---------------------------------------------------------------------------
// Kernel 2: global min/max reduction (float4 loads, wave32 shfl + block LDS)
// ---------------------------------------------------------------------------
__global__ __launch_bounds__(256) void coll_minmax(const float4* __restrict__ x4,
                                                   int n4, unsigned* ws) {
  unsigned lmin = 0xFFFFFFFFu, lmax = 0u;
  int stride = gridDim.x * blockDim.x;
  for (int i = blockIdx.x * blockDim.x + threadIdx.x; i < n4; i += stride) {
    float4 v = x4[i];
    unsigned k0 = enc_key(v.x), k1 = enc_key(v.y);
    unsigned k2 = enc_key(v.z), k3 = enc_key(v.w);
    unsigned kmin01 = k0 < k1 ? k0 : k1;
    unsigned kmax01 = k0 > k1 ? k0 : k1;
    unsigned kmin23 = k2 < k3 ? k2 : k3;
    unsigned kmax23 = k2 > k3 ? k2 : k3;
    unsigned kmin = kmin01 < kmin23 ? kmin01 : kmin23;
    unsigned kmax = kmax01 > kmax23 ? kmax01 : kmax23;
    lmin = lmin < kmin ? lmin : kmin;
    lmax = lmax > kmax ? lmax : kmax;
  }
  #pragma unroll
  for (int o = 16; o > 0; o >>= 1) {
    unsigned omin = __shfl_xor(lmin, o, 32);
    unsigned omax = __shfl_xor(lmax, o, 32);
    lmin = lmin < omin ? lmin : omin;
    lmax = lmax > omax ? lmax : omax;
  }
  __shared__ unsigned smin[8], smax[8];
  int lane = threadIdx.x & 31, wid = threadIdx.x >> 5;
  if (lane == 0) { smin[wid] = lmin; smax[wid] = lmax; }
  __syncthreads();
  if (threadIdx.x == 0) {
    unsigned bmin = smin[0], bmax = smax[0];
    #pragma unroll
    for (int wv = 1; wv < 8; ++wv) {
      bmin = bmin < smin[wv] ? bmin : smin[wv];
      bmax = bmax > smax[wv] ? bmax : smax[wv];
    }
    atomicMin(&ws[0], bmin);
    atomicMax(&ws[1], bmax);
  }
}

// ---------------------------------------------------------------------------
// Kernel 3: fused table T[m*9 + t] = co_flat[m] * w[t]  (m = ip*8+iq, 0..63)
// Rank-1 matmul on the WMMA unit: D[16x16] = A[16x4(K=1)] x B[4x16], 4 chunks.
// ---------------------------------------------------------------------------
#if __has_builtin(__builtin_amdgcn_wmma_f32_16x16x4_f32)
#define HAVE_WMMA_F32X4 1
#endif

__global__ __launch_bounds__(32) void coll_fuse_table(const float* __restrict__ co,
                                                      const float* __restrict__ w,
                                                      float* __restrict__ T) {
  int lane = threadIdx.x;     // 0..31
  int half = lane >> 4;       // 0: K rows {0,1}, 1: K rows {2,3}
  int l16  = lane & 15;

#if HAVE_WMMA_F32X4
  float wn = (l16 < 9) ? w[l16] : 0.0f;
  v2f b;
  b.x = (half == 0) ? wn : 0.0f;  // B[K=0][n] (half1 holds K=2 -> 0)
  b.y = 0.0f;                     // K=1 / K=3
  #pragma unroll
  for (int r = 0; r < 4; ++r) {
    v2f a;
    a.x = (half == 0) ? co[r * 16 + l16] : 0.0f;  // A[m][K=0] = co_flat
    a.y = 0.0f;
    v8f c = {};
    c = __builtin_amdgcn_wmma_f32_16x16x4_f32(
        false, a, false, b, (short)0, c, false, false);
    if (l16 < 9) {
      #pragma unroll
      for (int i = 0; i < 8; ++i) {
        int m = r * 16 + i + 8 * half;
        T[m * 9 + l16] = c[i];
      }
    }
  }
#else
  _Float16 wn = (_Float16)((l16 < 9) ? w[l16] : 0.0f);
  v16h b = {};
  if (half == 0) b[0] = wn;
  #pragma unroll
  for (int r = 0; r < 4; ++r) {
    v16h a = {};
    if (half == 0) a[0] = (_Float16)co[r * 16 + l16];
    v8f c = {};
    c = __builtin_amdgcn_wmma_f32_16x16x32_f16(
        false, a, false, b, (short)0, c, false, false);
    if (l16 < 9) {
      #pragma unroll
      for (int i = 0; i < 8; ++i) {
        int m = r * 16 + i + 8 * half;
        T[m * 9 + l16] = c[i];
      }
    }
  }
#endif
}

// ---------------------------------------------------------------------------
// Kernel 4: main stencil. One block = 8x8 pixel tile x 64 channels of one image.
// Halo tile (10x10x64 f32 = 25.6KB) staged into LDS with async-to-LDS
// (b128 preferred, b32 fallback), compute vectorized over channel quads.
// ---------------------------------------------------------------------------
#if __has_builtin(__builtin_amdgcn_global_load_async_to_lds_b128)
#define ASYNC128 1
typedef __attribute__((address_space(1))) v4i GAS_V4I;
typedef __attribute__((address_space(3))) v4i LAS_V4I;
#elif __has_builtin(__builtin_amdgcn_global_load_async_to_lds_b32)
#define ASYNC32 1
#endif
#if __has_builtin(__builtin_amdgcn_global_load_async_to_lds_b32)
typedef __attribute__((address_space(1))) int GAS_I;
typedef __attribute__((address_space(3))) int LAS_I;
#endif

__device__ __forceinline__ int bin_of(float v, float xmin, float scale) {
  int b = (int)floorf((v - xmin) * scale);
  b = b < 0 ? 0 : b;
  return b > (NB - 1) ? (NB - 1) : b;
}

__global__ __launch_bounds__(256) void coll_main(const float* __restrict__ x,
                                                 const float* __restrict__ Tg,
                                                 const unsigned* __restrict__ keys,
                                                 float* __restrict__ y) {
  __shared__ float xs[HALO_ELEMS];   // 25600 B
  __shared__ float Ts[64 * 9];       // 2304 B

  const float xmin  = dec_key(keys[0]);
  const float xmax  = dec_key(keys[1]);
  const float scale = (float)NB / (xmax - xmin + 1e-8f);

  const int tid = threadIdx.x;

  // stage fused table
  for (int i = tid; i < 64 * 9; i += 256) Ts[i] = Tg[i];

  // tile coordinates
  const int tilesW = DIMW / TW, tilesH = DIMH / TH;
  int b  = blockIdx.x;
  int tw = b % tilesW;
  int th = (b / tilesW) % tilesH;
  int n  = b / (tilesW * tilesH);
  const int h0 = th * TH - 1;
  const int w0 = tw * TW - 1;
  const size_t imgBase = (size_t)n * DIMH * DIMW * DIMC;

  // ---- stage halo tile into LDS in 16B quads (coalesced: c fastest) ----
  const float4 zero4 = make_float4(0.f, 0.f, 0.f, 0.f);
  for (int e4 = tid; e4 < HALO_ELEMS / 4; e4 += 256) {
    int q   = e4 & 15;          // channel quad 0..15  (c = q*4)
    int pix = e4 >> 4;          // 0..99 halo pixel
    int ww  = pix % HALO_W;
    int hh  = pix / HALO_W;
    int gh = h0 + hh, gw = w0 + ww;
    bool inb = (gh >= 0) & (gh < DIMH) & (gw >= 0) & (gw < DIMW);
    float* lp = &xs[e4 * 4];
    if (inb) {
      const float* gp = x + imgBase + ((size_t)gh * DIMW + gw) * DIMC + q * 4;
#if ASYNC128
      __builtin_amdgcn_global_load_async_to_lds_b128((GAS_V4I*)gp, (LAS_V4I*)lp, 0, 0);
#elif ASYNC32
      __builtin_amdgcn_global_load_async_to_lds_b32((GAS_I*)(gp + 0), (LAS_I*)(lp + 0), 0, 0);
      __builtin_amdgcn_global_load_async_to_lds_b32((GAS_I*)(gp + 1), (LAS_I*)(lp + 1), 0, 0);
      __builtin_amdgcn_global_load_async_to_lds_b32((GAS_I*)(gp + 2), (LAS_I*)(lp + 2), 0, 0);
      __builtin_amdgcn_global_load_async_to_lds_b32((GAS_I*)(gp + 3), (LAS_I*)(lp + 3), 0, 0);
#else
      *(float4*)lp = *(const float4*)gp;
#endif
    } else {
      *(float4*)lp = zero4;     // zero padding ("SAME" conv)
    }
  }
#if defined(ASYNC128) || defined(ASYNC32)
#if __has_builtin(__builtin_amdgcn_s_wait_asynccnt)
  __builtin_amdgcn_s_wait_asynccnt(0);
#else
  asm volatile("s_wait_asynccnt 0" ::: "memory");
#endif
#endif
  __syncthreads();

  // ---- compute: thread owns channel quad c4, pixels p with p%16 == tid/16 ----
  const int c4 = (tid & 15) * 4;   // channel base
  const int pg = tid >> 4;         // 0..15

  #pragma unroll
  for (int j = 0; j < 4; ++j) {
    int p  = j * 16 + pg;          // 0..63 within 8x8 tile
    int ph = p >> 3, pw = p & 7;
    int base = ((ph + 1) * HALO_W + (pw + 1)) * DIMC + c4;

    float4 xc = *(const float4*)&xs[base];
    int ip0 = bin_of(xc.x, xmin, scale);
    int ip1 = bin_of(xc.y, xmin, scale);
    int ip2 = bin_of(xc.z, xmin, scale);
    int ip3 = bin_of(xc.w, xmin, scale);
    const float* T0 = Ts + ip0 * (NB * 9);
    const float* T1 = Ts + ip1 * (NB * 9);
    const float* T2 = Ts + ip2 * (NB * 9);
    const float* T3 = Ts + ip3 * (NB * 9);

    float4 acc = zero4;
    #pragma unroll
    for (int dy = -1; dy <= 1; ++dy) {
      #pragma unroll
      for (int dx = -1; dx <= 1; ++dx) {
        int    t  = (dy + 1) * 3 + (dx + 1);
        float4 xq = *(const float4*)&xs[base + (dy * HALO_W + dx) * DIMC];
        int iq0 = bin_of(xq.x, xmin, scale);
        int iq1 = bin_of(xq.y, xmin, scale);
        int iq2 = bin_of(xq.z, xmin, scale);
        int iq3 = bin_of(xq.w, xmin, scale);
        acc.x = fmaf(xq.x, T0[iq0 * 9 + t], acc.x);
        acc.y = fmaf(xq.y, T1[iq1 * 9 + t], acc.y);
        acc.z = fmaf(xq.z, T2[iq2 * 9 + t], acc.z);
        acc.w = fmaf(xq.w, T3[iq3 * 9 + t], acc.w);
      }
    }

    int gh = th * TH + ph, gw = tw * TW + pw;
    *(float4*)&y[imgBase + ((size_t)gh * DIMW + gw) * DIMC + c4] = acc;
  }
}

// ---------------------------------------------------------------------------
// Launch
// ---------------------------------------------------------------------------
extern "C" void kernel_launch(void* const* d_in, const int* in_sizes, int n_in,
                              void* d_out, int out_size, void* d_ws, size_t ws_size,
                              hipStream_t stream) {
  const float* x  = (const float*)d_in[0];
  const float* co = (const float*)d_in[1];
  const float* w  = (const float*)d_in[2];
  float* y        = (float*)d_out;

  unsigned* keys = (unsigned*)d_ws;
  float*    T    = (float*)((char*)d_ws + 64);  // 576 floats

  const int n = in_sizes[0];  // 8*128*128*64 (divisible by 4)

  coll_init_ws<<<1, 1, 0, stream>>>(keys);
  coll_minmax<<<1024, 256, 0, stream>>>((const float4*)x, n / 4, keys);
  coll_fuse_table<<<1, 32, 0, stream>>>(co, w, T);

  const int blocks = DIMN * (DIMH / TH) * (DIMW / TW);  // 2048
  coll_main<<<blocks, 256, 0, stream>>>(x, T, keys, y);
}
